// LinearAttention_77532749627908
// MI455X (gfx1250) — matrix operand order repaired and checked by reading, not compiled, not run on hardware
//
#include <hip/hip_runtime.h>

typedef __bf16 bf16;
typedef __attribute__((ext_vector_type(16))) __bf16 v16bf;
typedef __attribute__((ext_vector_type(8)))  float  v8f;

#define DIMC   256
#define NPIX   4096
#define NH     8
#define DH     64
#define HIDDEN 512
#define INV_SCALE 0.125f   // 1/sqrt(64)

// ---------------- WMMA helpers (wave32, v_wmma_f32_16x16x32_bf16) ----------------

__device__ __forceinline__ int lane_id() { return threadIdx.x & 31; }

__device__ __forceinline__ v8f wmma_bf16(v16bf a, v16bf b, v8f c) {
  return __builtin_amdgcn_wmma_f32_16x16x32_bf16(false, a, false, b, (short)0, c,
                                                 false, false);
}

// A fragment (16x32) from row-major bf16 [M][pitch] (LDS or global), rows 16B aligned.
// CDNA5 A layout: lane<16 -> K 0..7 & 16..23, lane>=16 -> K 8..15 & 24..31.
__device__ __forceinline__ v16bf load_a16(const bf16* base, int pitch, int m0, int k0) {
  int l = lane_id() & 15, hi = lane_id() >> 4;
  const bf16* p = base + (size_t)(m0 + l) * pitch + k0 + hi * 8;
  union { uint4 q[2]; v16bf v; } u;
  u.q[0] = *(const uint4*)p;
  u.q[1] = *(const uint4*)(p + 16);
  return u.v;
}

// B fragment (32x16) from LDS stored TRANSPOSED: base[n][k] (n-major rows).
// CDNA5 B layout: lanes 0-15 K=0..7 & 16..23, lanes 16-31 K=8..15 & 24..31, N = lane&15.
__device__ __forceinline__ v16bf load_b_lds(const bf16* base, int pitch, int k0, int n0) {
  int l = lane_id() & 15, hi = lane_id() >> 4;
  const bf16* p = base + (n0 + l) * pitch + k0 + hi * 8;
  union { uint4 q[2]; v16bf v; } u;
  u.q[0] = *(const uint4*)p;
  u.q[1] = *(const uint4*)(p + 16);
  return u.v;
}

// Store f32 D tile (16x16) as bf16 to LDS row-major [M][pitch].
__device__ __forceinline__ void store_d_rowmajor(bf16* base, int pitch, int m0, int n0, v8f c) {
  int l = lane_id() & 15, hi = lane_id() >> 4;
#pragma unroll
  for (int r = 0; r < 8; ++r)
    base[(m0 + r + 8 * hi) * pitch + n0 + l] = (bf16)c[r];
}

// Store f32 D tile transposed (as [n][m]) to LDS.
__device__ __forceinline__ void store_d_transposed(bf16* base, int pitch, int m0, int n0, v8f c) {
  int l = lane_id() & 15, hi = lane_id() >> 4;
#pragma unroll
  for (int r = 0; r < 8; ++r)
    base[(n0 + l) * pitch + m0 + r + 8 * hi] = (bf16)c[r];
}

// ---------------- Kernel 0: one-shot weight conversion f32 -> bf16 ---------------

#define WQKV_ELEMS (3 * HIDDEN * DIMC)   // 393216
#define WOUT_ELEMS (DIMC * HIDDEN)       // 131072

__global__ void convert_weights(const float* __restrict__ w_qkv,
                                const float* __restrict__ w_out,
                                bf16* __restrict__ wqkv_b,
                                bf16* __restrict__ wout_b) {
  int i = blockIdx.x * blockDim.x + threadIdx.x;
  if (i < WQKV_ELEMS) {
    wqkv_b[i] = (bf16)w_qkv[i];
  } else {
    int j = i - WQKV_ELEMS;
    if (j < WOUT_ELEMS) wout_b[j] = (bf16)w_out[j];
  }
}

// ---------------- Kernel 1: fused KV generation + ctx = K @ V^T (partials) -------

#define K1_XSP 272   // xs pitch: [128 px][256 ch + pad] -> 136-dword rows (8-bank stride)
#define K1_KVP 136   // kv pitch: [128 rows][128 px + pad]

__global__ __launch_bounds__(256) void ctx_kernel(const float* __restrict__ x,
                                                  const bf16* __restrict__ wqkv_b,
                                                  float* __restrict__ part) {
  __shared__ __align__(16) bf16 xs[128 * K1_XSP];   // X tile, stored [pixel][chan]
  __shared__ __align__(16) bf16 kv[128 * K1_KVP];   // rows 0..63 = K, 64..127 = V

  int chunk = blockIdx.x & 7;        // 8 chunks of 512 pixels
  int bh    = blockIdx.x >> 3;       // 0..127
  int b = bh >> 3, h = bh & 7;
  int wave = threadIdx.x >> 5;       // 0..7
  const float* xb = x + (size_t)b * DIMC * NPIX;

  // this wave produces kv rows [wave*16, wave*16+16)
  int mb = wave;
  int grow0 = (mb < 4) ? (HIDDEN + h * DH + mb * 16)
                       : (2 * HIDDEN + h * DH + (mb - 4) * 16);
  const bf16* wrow = wqkv_b + (size_t)grow0 * DIMC;

  // Preload the 8 A fragments of Wkv (constant across sections)
  v16bf wa[8];
#pragma unroll
  for (int kk = 0; kk < 8; ++kk) wa[kk] = load_a16(wrow, DIMC, 0, kk * 32);

  // ctx accumulators: wave owns subtiles (di, ej0) and (di, ej0+1)
  int di  = wave & 3;
  int ej0 = (wave >> 2) * 2;
  v8f cacc[2] = {};

  for (int s = 0; s < 4; ++s) {
    int p0 = chunk * 512 + s * 128;
    __syncthreads();                         // protect xs/kv reuse
    // stage x[0:256][p0:p0+128] -> xs[p][c], packed pair stores (b32)
    {
      int pl = threadIdx.x & 127;
      int ch = threadIdx.x >> 7;             // 0..1
      const float* xc = xb + p0 + pl;
#pragma unroll 4
      for (int cc = 0; cc < 64; ++cc) {
        int c = ch * 128 + cc * 2;
        union { bf16 v[2]; unsigned u; } pk;
        pk.v[0] = (bf16)xc[(size_t)c * NPIX];
        pk.v[1] = (bf16)xc[(size_t)(c + 1) * NPIX];
        *(unsigned*)&xs[pl * K1_XSP + c] = pk.u;
      }
    }
    __syncthreads();
    // phase 1: kv[mb*16..][0..127] = Wkv_rows @ X  (K = 256 channels)
    {
      v8f acc[8] = {};
#pragma unroll
      for (int kk = 0; kk < 8; ++kk) {
        int k0 = kk * 32;
#pragma unroll
        for (int nb = 0; nb < 8; ++nb) {
          v16bf bb = load_b_lds(xs, K1_XSP, k0, nb * 16);
          acc[nb] = wmma_bf16(wa[kk], bb, acc[nb]);
        }
      }
#pragma unroll
      for (int nb = 0; nb < 8; ++nb)
        store_d_rowmajor(kv, K1_KVP, mb * 16, nb * 16, acc[nb]);
    }
    __syncthreads();
    // phase 2: ctx += K @ V^T over this section's 128 pixels
#pragma unroll
    for (int nn = 0; nn < 4; ++nn) {
      int n0 = nn * 32;
      v16bf a = load_a16(kv, K1_KVP, di * 16, n0);
#pragma unroll
      for (int t = 0; t < 2; ++t) {
        v16bf bb = load_b_lds(kv + 64 * K1_KVP, K1_KVP, n0, (ej0 + t) * 16);
        cacc[t] = wmma_bf16(a, bb, cacc[t]);
      }
    }
  }

  // deterministic partial write: part[chunk][bh][d][e]  (non-temporal: not re-read soon,
  // and keeping x resident in the 192MB L2 between passes is worth more)
  int l = lane_id() & 15, hi = lane_id() >> 4;
  float* pb = part + ((size_t)chunk * 128 + bh) * DH * DH;
#pragma unroll
  for (int t = 0; t < 2; ++t)
#pragma unroll
    for (int r = 0; r < 8; ++r) {
      int d = di * 16 + r + 8 * hi;
      int e = (ej0 + t) * 16 + l;
      __builtin_nontemporal_store(cacc[t][r], &pb[d * DH + e]);
    }
}

// ---------------- Kernel 1b: reduce partials, fold 1/sqrt(d) scale --------------

__global__ void reduce_ctx(const float* __restrict__ part, float* __restrict__ ctx) {
  int i = blockIdx.x * blockDim.x + threadIdx.x;   // 0 .. 128*64*64-1
  float s = 0.f;
#pragma unroll
  for (int c = 0; c < 8; ++c) s += part[(size_t)c * (128 * DH * DH) + i];
  ctx[i] = s * INV_SCALE;
}

// ---------------- Kernel 2: fused Q gen + att = ctx^T@Q + out = Wout@att + b ----

#define K2_XSP 272   // xs:   [64 px][256 ch + pad]
#define K2_QSP 72    // qs:   [64 px][64 d + pad]    (stored transposed for B-frags)
#define K2_CTP 72    // cts:  [64 e][64 d + pad]     (ctx^T, A-frags)
#define K2_ATP 528   // atts: [64 px][512 e + pad]   (stored transposed for B-frags)

__global__ __launch_bounds__(256) void out_kernel(const float* __restrict__ x,
                                                  const bf16* __restrict__ wqkv_b,
                                                  const bf16* __restrict__ wout_b,
                                                  const float* __restrict__ b_out,
                                                  const float* __restrict__ ctx,
                                                  float* __restrict__ out) {
  __shared__ __align__(16) bf16 xs[64 * K2_XSP];
  __shared__ __align__(16) bf16 qs[64 * K2_QSP];
  __shared__ __align__(16) bf16 cts[64 * K2_CTP];
  __shared__ __align__(16) bf16 atts[64 * K2_ATP];

  int tile = blockIdx.x & 63;        // 64 tiles of 64 pixels
  int b    = blockIdx.x >> 6;
  int p0   = tile * 64;
  int wave = threadIdx.x >> 5;
  const float* xb = x + (size_t)b * DIMC * NPIX;

  // stage x[0:256][p0:p0+64] -> xs[p][c]
  {
    int pl    = threadIdx.x & 63;
    int cpart = threadIdx.x >> 6;    // 0..3
    const float* xc = xb + p0 + pl;
#pragma unroll 4
    for (int cc = 0; cc < 32; ++cc) {
      int c = cpart * 64 + cc * 2;
      union { bf16 v[2]; unsigned u; } pk;
      pk.v[0] = (bf16)xc[(size_t)c * NPIX];
      pk.v[1] = (bf16)xc[(size_t)(c + 1) * NPIX];
      *(unsigned*)&xs[pl * K2_XSP + c] = pk.u;
    }
  }

  int mb  = wave & 3;                // 16-row block (M of q / att GEMMs)
  int nb0 = (wave >> 2) * 2;         // 2 pixel blocks per wave (N = 64 -> 4 blocks)

  for (int h = 0; h < NH; ++h) {
    __syncthreads();                 // first iter: xs ready; later: qs/cts reuse
    // stage ctx^T for this head: cts[e][d] = ctx[bh][d][e] (already scaled)
    {
      int e     = threadIdx.x & 63;
      int dpart = threadIdx.x >> 6;
      const float* cg = ctx + (size_t)(b * NH + h) * DH * DH;
#pragma unroll
      for (int i = 0; i < 16; ++i) {
        int d = dpart * 16 + i;
        cts[e * K2_CTP + d] = (bf16)cg[d * DH + e];
      }
    }
    // Q GEMM: q[64][64] = Wq_head @ Xtile  (K = 256); store transposed to qs[p][d]
    {
      const bf16* wq = wqkv_b + (size_t)(h * DH + mb * 16) * DIMC;
      v8f acc[2] = {};
#pragma unroll
      for (int kk = 0; kk < 8; ++kk) {
        int k0 = kk * 32;
        v16bf a = load_a16(wq, DIMC, 0, k0);
#pragma unroll
        for (int t = 0; t < 2; ++t) {
          v16bf bb = load_b_lds(xs, K2_XSP, k0, (nb0 + t) * 16);
          acc[t] = wmma_bf16(a, bb, acc[t]);
        }
      }
#pragma unroll
      for (int t = 0; t < 2; ++t)
        store_d_transposed(qs, K2_QSP, mb * 16, (nb0 + t) * 16, acc[t]);
    }
    __syncthreads();
    // att GEMM: att[e=64][p=64] = ctx^T @ q  (K = 64)
    {
      v8f acc[2] = {};
#pragma unroll
      for (int kk = 0; kk < 2; ++kk) {
        int k0 = kk * 32;
        v16bf a = load_a16(cts, K2_CTP, mb * 16, k0);
#pragma unroll
        for (int t = 0; t < 2; ++t) {
          v16bf bb = load_b_lds(qs, K2_QSP, k0, (nb0 + t) * 16);
          acc[t] = wmma_bf16(a, bb, acc[t]);
        }
      }
#pragma unroll
      for (int t = 0; t < 2; ++t)
        store_d_transposed(atts, K2_ATP, h * DH + mb * 16, (nb0 + t) * 16, acc[t]);
    }
  }
  __syncthreads();

  // Final GEMM: out[256][64] = Wout @ att + bias  (K = 512); two 128-row halves
  int l = lane_id() & 15, hi = lane_id() >> 4;
#pragma unroll
  for (int oh = 0; oh < 2; ++oh) {
    const bf16* wo = wout_b + (size_t)(oh * 128 + wave * 16) * HIDDEN;
    v8f acc[4] = {};
#pragma unroll
    for (int kk = 0; kk < 16; ++kk) {
      int k0 = kk * 32;
      v16bf a = load_a16(wo, HIDDEN, 0, k0);
#pragma unroll
      for (int nb = 0; nb < 4; ++nb) {
        v16bf bb = load_b_lds(atts, K2_ATP, k0, nb * 16);
        acc[nb] = wmma_bf16(a, bb, acc[nb]);
      }
    }
#pragma unroll
    for (int nb = 0; nb < 4; ++nb)
#pragma unroll
      for (int r = 0; r < 8; ++r) {
        int o = oh * 128 + wave * 16 + r + 8 * hi;
        int p = nb * 16 + l;
        // non-temporal: output is never re-read; don't evict x/weights from L2
        __builtin_nontemporal_store(acc[nb][r] + b_out[o],
                                    &out[((size_t)b * DIMC + o) * NPIX + p0 + p]);
      }
  }
}

// ---------------- Launch ---------------------------------------------------------

extern "C" void kernel_launch(void* const* d_in, const int* in_sizes, int n_in,
                              void* d_out, int out_size, void* d_ws, size_t ws_size,
                              hipStream_t stream) {
  const float* x     = (const float*)d_in[0];
  const float* w_qkv = (const float*)d_in[1];
  const float* w_out = (const float*)d_in[2];
  const float* b_out = (const float*)d_in[3];
  float* out = (float*)d_out;

  // workspace layout:
  //   part   : [8][128][64][64] f32 partials (16 MB)
  //   ctx    : [128][64][64] f32 (2 MB)
  //   wqkv_b : [1536][256] bf16 (768 KB)
  //   wout_b : [256][512] bf16 (256 KB)
  float* part  = (float*)d_ws;
  float* ctx   = part + (size_t)8 * 128 * DH * DH;
  bf16*  wqkvb = (bf16*)(ctx + (size_t)128 * DH * DH);
  bf16*  woutb = wqkvb + WQKV_ELEMS;

  convert_weights<<<(WQKV_ELEMS + WOUT_ELEMS) / 256, 256, 0, stream>>>(
      w_qkv, w_out, wqkvb, woutb);
  ctx_kernel<<<1024, 256, 0, stream>>>(x, wqkvb, part);
  reduce_ctx<<<(128 * DH * DH) / 256, 256, 0, stream>>>(part, ctx);
  out_kernel<<<1024, 256, 0, stream>>>(x, wqkvb, woutb, b_out, ctx, out);
}
